// RNNBase_23974507446904
// MI455X (gfx1250) — matrix operand order repaired and checked by reading, not compiled, not run on hardware
//
#include <hip/hip_runtime.h>

// ---------------- problem constants ----------------
#define B_ 64
#define T_ 256
#define D_ 512
#define H_ 1024
#define G_ 4096   // 4*H gates

typedef __attribute__((ext_vector_type(16))) __bf16        v16bf;
typedef __attribute__((ext_vector_type(8)))  float         v8f;
typedef __attribute__((ext_vector_type(4)))  unsigned int  u32x4;

union FragU { u32x4 u[2]; v16bf v; };

__device__ __forceinline__ unsigned short f2bf(float f) {
  union { float f; unsigned u; } v; v.f = f;
  unsigned u = v.u;
  return (unsigned short)((u + 0x7FFFu + ((u >> 16) & 1u)) >> 16);  // RNE
}

// ---------------- packing kernels ----------------
// Pack W [G, K] (row-major, gate-major rows: q*H + u) into WMMA-B tiles with
// gate-interleaved column permutation: column c <-> original row (c%4)*H + c/4.
// Tile (nt, kt) holds 512 bf16 in per-lane fragment order:
//   elem index = tile*512 + lane*16 + i, value = W[perm(nt*16 + lane&15)][kt*32 + (lane>>4)*16 + i]
__global__ void pack_w(const float* __restrict__ W, unsigned short* __restrict__ out, int K) {
  int idx = blockIdx.x * blockDim.x + threadIdx.x;
  if (idx >= G_ * K) return;
  int i    = idx & 15;
  int l    = (idx >> 4) & 31;
  int tile = idx >> 9;
  int ktiles = K >> 5;
  int kt = tile % ktiles;
  int nt = tile / ktiles;
  int c   = nt * 16 + (l & 15);
  int row = (c & 3) * H_ + (c >> 2);
  int k   = kt * 32 + (l >> 4) * 16 + i;
  out[idx] = f2bf(W[(size_t)row * K + k]);
}

__global__ void pack_bias(const float* __restrict__ bi, const float* __restrict__ bh,
                          float* __restrict__ out) {
  int c = blockIdx.x * blockDim.x + threadIdx.x;
  if (c >= G_) return;
  int row = (c & 3) * H_ + (c >> 2);
  out[c] = bi[row] + bh[row];
}

__global__ void to_bf16_k(const float* __restrict__ in, unsigned short* __restrict__ out, int n) {
  int i = blockIdx.x * blockDim.x + threadIdx.x;
  if (i < n) out[i] = f2bf(in[i]);
}

__global__ void init_state(unsigned short* __restrict__ h, float* __restrict__ c) {
  int i = blockIdx.x * blockDim.x + threadIdx.x;
  if (i < 2 * 2 * B_ * H_) h[i] = 0;        // both parities, both dirs
  if (i < 2 * B_ * H_)     c[i] = 0.f;      // both dirs
}

__global__ void copy_last(const float* __restrict__ src, float* __restrict__ dst, int n) {
  int i = blockIdx.x * blockDim.x + threadIdx.x;
  if (i < n) dst[i] = src[i];
}

// ---------------- fused LSTM step ----------------
// One wave computes a 16x16 gate tile: gates = bias + h_prev @ Whh^T + x_t @ Wih^T
// (columns gate-interleaved), then quad-shuffles i/f/g/o together and updates c/h.
// grid: 2048 waves = dir(2) x mtile(4) x ntile(256); 128 threads/block = 4 waves.
__global__ __launch_bounds__(128)
void lstm_step(const unsigned short* __restrict__ whh,   // [2][G_*H_] packed
               const unsigned short* __restrict__ wih,   // [2][G_*in_dim] packed
               const float* __restrict__ bias,           // [2][G_] permuted
               const unsigned short* __restrict__ xin,   // [B, T, in_dim] bf16
               const unsigned short* __restrict__ h_prev,// [2][B_][H_] bf16
               unsigned short* __restrict__ h_next,      // [2][B_][H_] bf16
               float* __restrict__ c_state,              // [2][B_][H_] f32
               unsigned short* __restrict__ out_bf,      // [B,T,2H] bf16 or null
               float* __restrict__ out_f,                // [B,T,2H] f32 or null
               int step, int in_dim)
{
  const int lane = threadIdx.x & 31;
  const int w  = blockIdx.x * (blockDim.x >> 5) + (threadIdx.x >> 5);
  const int d  = w >> 10;           // direction
  const int mt = (w >> 8) & 3;      // batch tile
  const int nt = w & 255;           // gate-column tile
  const int t  = d ? (T_ - 1 - step) : step;

  const int half  = lane >> 4;
  const int mrowA = mt * 16 + (lane & 15);   // A-matrix row (batch) for this lane
  const int ccol  = nt * 16 + (lane & 15);   // permuted gate column for B/D

  const float bval = bias[d * G_ + ccol];
  v8f acc;
  #pragma unroll
  for (int r = 0; r < 8; ++r) acc[r] = bval;

  // ---- recurrent contribution: A = h_prev[d] (K = H_) ----
  {
    const unsigned short* arow  = h_prev + ((size_t)(d * B_ + mrowA)) * H_;
    const unsigned short* bbase = whh + (size_t)d * G_ * H_
                                + (size_t)(nt * (H_ >> 5)) * 512 + lane * 16;
    #pragma unroll 4
    for (int kt = 0; kt < (H_ >> 5); ++kt) {
      FragU a, b;
      const u32x4* ap = (const u32x4*)(arow + kt * 32 + half * 16);
      a.u[0] = ap[0];           // K = kt*32 + half*16 .. +7
      a.u[1] = ap[1];           // K = kt*32 + half*16 + 8 .. +15
      const u32x4* bp = (const u32x4*)(bbase + (size_t)kt * 512);
      b.u[0] = bp[0];
      b.u[1] = bp[1];
      acc = __builtin_amdgcn_wmma_f32_16x16x32_bf16(false, a.v, false, b.v,
                                                    (short)0, acc, false, false);
    }
  }

  // ---- input contribution: A = x_t rows (K = in_dim) ----
  {
    const unsigned short* arow  = xin + ((size_t)mrowA * T_ + t) * in_dim;
    const unsigned short* bbase = wih + (size_t)d * G_ * in_dim
                                + (size_t)(nt * (in_dim >> 5)) * 512 + lane * 16;
    #pragma unroll 4
    for (int kt = 0; kt < (in_dim >> 5); ++kt) {
      FragU a, b;
      const u32x4* ap = (const u32x4*)(arow + kt * 32 + half * 16);
      a.u[0] = ap[0];
      a.u[1] = ap[1];
      const u32x4* bp = (const u32x4*)(bbase + (size_t)kt * 512);
      b.u[0] = bp[0];
      b.u[1] = bp[1];
      acc = __builtin_amdgcn_wmma_f32_16x16x32_bf16(false, a.v, false, b.v,
                                                    (short)0, acc, false, false);
    }
  }

  // ---- gate combine + pointwise update ----
  // D tile: element (M = r + 8*half, N = lane&15). Columns 4u..4u+3 hold
  // gates i,f,g,o of unit u = nt*4 + (lane&15)/4. Quad leader does the update.
  const int  qbase  = lane & ~3;
  const bool leader = (lane & 3) == 0;
  const int  unit   = ccol >> 2;             // exact for leader lanes
  #pragma unroll
  for (int r = 0; r < 8; ++r) {
    float dv = acc[r];
    float iv = __shfl(dv, qbase + 0, 32);
    float fv = __shfl(dv, qbase + 1, 32);
    float gv = __shfl(dv, qbase + 2, 32);
    float ov = __shfl(dv, qbase + 3, 32);
    if (leader) {
      int brow = mt * 16 + r + 8 * half;     // batch row of this D element
      size_t sidx = (size_t)(d * B_ + brow) * H_ + unit;
      float cold = c_state[sidx];
      float si = 1.f / (1.f + __expf(-iv));
      float sf = 1.f / (1.f + __expf(-fv));
      float so = 1.f / (1.f + __expf(-ov));
      float tg = 2.f / (1.f + __expf(-2.f * gv)) - 1.f;   // tanh(g)
      float cn = sf * cold + si * tg;
      float th = 2.f / (1.f + __expf(-2.f * cn)) - 1.f;   // tanh(c)
      float hv = so * th;
      c_state[sidx] = cn;
      h_next[sidx]  = f2bf(hv);
      size_t oidx = ((size_t)brow * T_ + t) * (2 * H_) + (size_t)d * H_ + unit;
      if (out_bf) out_bf[oidx] = f2bf(hv);
      if (out_f)  out_f[oidx]  = hv;
    }
  }
}

// ---------------- host launcher ----------------
extern "C" void kernel_launch(void* const* d_in, const int* in_sizes, int n_in,
                              void* d_out, int out_size, void* d_ws, size_t ws_size,
                              hipStream_t stream) {
  (void)in_sizes; (void)n_in; (void)out_size; (void)ws_size;

  const float* x = (const float*)d_in[0];
  // dict order: x, then for dir in (f,b): for layer in (0,1): w_ih, w_hh, b_ih, b_hh
  const float* w_ih[2][2]; const float* w_hh[2][2];
  const float* b_ih[2][2]; const float* b_hh[2][2];
  for (int dir = 0; dir < 2; ++dir)
    for (int layer = 0; layer < 2; ++layer) {
      int base = 1 + dir * 8 + layer * 4;
      w_ih[layer][dir] = (const float*)d_in[base + 0];
      w_hh[layer][dir] = (const float*)d_in[base + 1];
      b_ih[layer][dir] = (const float*)d_in[base + 2];
      b_hh[layer][dir] = (const float*)d_in[base + 3];
    }

  // workspace carve-up (~153 MB)
  char* ws = (char*)d_ws;
  unsigned short* xbf     = (unsigned short*)(ws + 0);            // 16 MB
  unsigned short* whh_pk  = (unsigned short*)(ws + 16777216);     // 32 MB [layer][dir]*G_*H_
  unsigned short* wih0_pk = (unsigned short*)(ws + 50331648);     //  8 MB [dir]*G_*512
  unsigned short* wih1_pk = (unsigned short*)(ws + 58720256);     // 32 MB [dir]*G_*2048
  float*          bias_pk = (float*)        (ws + 92274688);      // 64 KB [layer][dir][G_]
  unsigned short* hbuf    = (unsigned short*)(ws + 92340224);     // 512 KB [parity][dir][B][H]
  float*          cbuf    = (float*)        (ws + 92864512);      // 512 KB [dir][B][H]
  unsigned short* out0bf  = (unsigned short*)(ws + 93388800);     // 64 MB [B,T,2H]

  // 1) convert x to bf16
  to_bf16_k<<<(B_ * T_ * D_ + 255) / 256, 256, 0, stream>>>(x, xbf, B_ * T_ * D_);

  // 2) pack weights / biases
  for (int layer = 0; layer < 2; ++layer)
    for (int dir = 0; dir < 2; ++dir) {
      pack_w<<<(G_ * H_ + 255) / 256, 256, 0, stream>>>(
          w_hh[layer][dir], whh_pk + (size_t)(layer * 2 + dir) * G_ * H_, H_);
      int K = layer == 0 ? D_ : 2 * H_;
      unsigned short* dst = layer == 0 ? wih0_pk + (size_t)dir * G_ * D_
                                       : wih1_pk + (size_t)dir * G_ * 2 * H_;
      pack_w<<<(G_ * K + 255) / 256, 256, 0, stream>>>(w_ih[layer][dir], dst, K);
      pack_bias<<<(G_ + 255) / 256, 256, 0, stream>>>(
          b_ih[layer][dir], b_hh[layer][dir], bias_pk + (size_t)(layer * 2 + dir) * G_);
    }

  const int hpar = 2 * B_ * H_;  // parity stride in elements (covers both dirs)

  // 3) layer 0 scan: input xbf (in_dim=512), output -> out0bf (bf16)
  init_state<<<(2 * hpar + 255) / 256, 256, 0, stream>>>(hbuf, cbuf);
  for (int s = 0; s < T_; ++s) {
    lstm_step<<<512, 128, 0, stream>>>(
        whh_pk, wih0_pk, bias_pk, xbf,
        hbuf + (s & 1) * hpar, hbuf + ((s + 1) & 1) * hpar, cbuf,
        out0bf, (float*)nullptr, s, D_);
  }

  // 4) layer 1 scan: input out0bf (in_dim=2048), output -> d_out (fp32)
  init_state<<<(2 * hpar + 255) / 256, 256, 0, stream>>>(hbuf, cbuf);
  for (int s = 0; s < T_; ++s) {
    lstm_step<<<512, 128, 0, stream>>>(
        whh_pk + (size_t)2 * G_ * H_, wih1_pk, bias_pk + (size_t)2 * G_, out0bf,
        hbuf + (s & 1) * hpar, hbuf + ((s + 1) & 1) * hpar, cbuf,
        (unsigned short*)nullptr, (float*)d_out, s, 2 * H_);
  }

  // 5) second tuple output: h[-1] = output rows of batch b = B-1
  const int nlast = T_ * 2 * H_;
  copy_last<<<(nlast + 255) / 256, 256, 0, stream>>>(
      (const float*)d_out + (size_t)(B_ - 1) * T_ * 2 * H_,
      (float*)d_out + (size_t)B_ * T_ * 2 * H_, nlast);
}
